// Transformer_15960098472429
// MI455X (gfx1250) — compile-verified, matrix operand order checked by reading, not run on hardware
//
#include <hip/hip_runtime.h>
#include <math.h>

// ---------------------------------------------------------------------------
// Transformer forward for MI455X (gfx1250): bf16 WMMA, f32 accumulate,
// async global->LDS double-buffered GEMM, pre-transposed bf16 weights.
// ---------------------------------------------------------------------------

typedef __attribute__((ext_vector_type(16))) __bf16 v16bf;
typedef __attribute__((ext_vector_type(8)))  __bf16 v8bf;
typedef __attribute__((ext_vector_type(8)))  float  v8f;
typedef __attribute__((ext_vector_type(4)))  int    v4i;

#define LAYERS 8
#define BATCH  4
#define SEQ    1024
#define DMODEL 1024
#define NHEAD  16
#define DHEAD  64
#define DFF    4096
#define ROWS   (BATCH * SEQ)   // 4096
#define LN_EPS 1e-5f

// WMMA 16-bit fragment K-index pattern (ISA 7.12.2):
// lanes 0-15: elems 0..7 -> K 0..7,  elems 8..15 -> K 16..23
// lanes16-31: elems 0..7 -> K 8..15, elems 8..15 -> K 24..31
#define FRAG_K(e, half) (((e) & 7) + ((half) << 3) + (((e) >> 3) << 4))

#if defined(__gfx1250__) && __has_builtin(__builtin_amdgcn_global_load_async_to_lds_b128)
#define USE_ASYNC 1
#else
#define USE_ASYNC 0
#endif

__device__ __forceinline__ void async_copy16(const void* g, void* l) {
#if USE_ASYNC
    __builtin_amdgcn_global_load_async_to_lds_b128(
        (__attribute__((address_space(1))) v4i*)g,
        (__attribute__((address_space(3))) v4i*)l, 0, 0);
#else
    *(v8bf*)l = *(const v8bf*)g;
#endif
}

__device__ __forceinline__ void wait_async_keep2() {
#if USE_ASYNC
#if __has_builtin(__builtin_amdgcn_s_wait_asynccnt)
    __builtin_amdgcn_s_wait_asynccnt(2);
#else
    asm volatile("s_wait_asynccnt 0x2" ::: "memory");
#endif
#endif
}

__device__ __forceinline__ void wait_async_zero() {
#if USE_ASYNC
#if __has_builtin(__builtin_amdgcn_s_wait_asynccnt)
    __builtin_amdgcn_s_wait_asynccnt(0);
#else
    asm volatile("s_wait_asynccnt 0x0" ::: "memory");
#endif
#endif
}

// ---------------------------------------------------------------------------
// Tiled transpose + f32->bf16:  in[l][K][N] (f32)  ->  out[l][N][K] (bf16)
// Coalesced on both sides via a 32x32 LDS tile. One-time cost per launch:
// ~576MB of traffic ~= 25us at 23.3 TB/s, noise vs the matmul work.
// ---------------------------------------------------------------------------
__global__ __launch_bounds__(256)
void transpose_to_bf16_kernel(const float* __restrict__ in,
                              __bf16* __restrict__ out, int K, int N) {
    __shared__ float tile[32][33];
    const int l  = blockIdx.z;
    const float* inp = in + (size_t)l * K * N;
    __bf16*     outp = out + (size_t)l * K * N;
    const int k0 = blockIdx.y * 32, n0 = blockIdx.x * 32;
    const int tx = threadIdx.x;          // 0..31
    const int ty = threadIdx.y;          // 0..7
    for (int yy = ty; yy < 32; yy += 8)
        tile[yy][tx] = inp[(size_t)(k0 + yy) * N + (n0 + tx)];
    __syncthreads();
    for (int yy = ty; yy < 32; yy += 8)
        outp[(size_t)(n0 + yy) * K + (k0 + tx)] = (__bf16)tile[tx][yy];
}

// ---------------------------------------------------------------------------
// Row LayerNorm over D=1024; optional f32 and bf16 outputs
// ---------------------------------------------------------------------------
__global__ __launch_bounds__(256)
void layernorm_kernel(const float* __restrict__ x, const float* __restrict__ g,
                      const float* __restrict__ bta,
                      float* __restrict__ outf, __bf16* __restrict__ outb) {
    __shared__ float s1[256], s2[256];
    const int row = blockIdx.x;
    const float* xr = x + (size_t)row * DMODEL;
    float a0 = 0.f, a1 = 0.f;
    for (int i = threadIdx.x; i < DMODEL; i += 256) {
        float v = xr[i];
        a0 += v; a1 += v * v;
    }
    s1[threadIdx.x] = a0; s2[threadIdx.x] = a1;
    __syncthreads();
    for (int s = 128; s > 0; s >>= 1) {
        if ((int)threadIdx.x < s) {
            s1[threadIdx.x] += s1[threadIdx.x + s];
            s2[threadIdx.x] += s2[threadIdx.x + s];
        }
        __syncthreads();
    }
    const float mu  = s1[0] * (1.0f / DMODEL);
    const float var = s2[0] * (1.0f / DMODEL) - mu * mu;
    const float r   = rsqrtf(var + LN_EPS);
    for (int i = threadIdx.x; i < DMODEL; i += 256) {
        float v = (xr[i] - mu) * r * g[i] + bta[i];
        if (outf) outf[(size_t)row * DMODEL + i] = v;
        if (outb) outb[(size_t)row * DMODEL + i] = (__bf16)v;
    }
}

// ---------------------------------------------------------------------------
// bf16 WMMA GEMM:  C[M,N] = A[M,K] @ Wt[N,K]^T + bias, fused gelu / residual.
// Block tile 128x128, BK=32, 8 waves in a 4(M) x 2(N) grid; each wave owns a
// 32x64 strip = 2x4 f32 accumulators. Both LDS tiles are filled with async
// b128 global->LDS copies (ASYNCcnt) and double-buffered.
// ---------------------------------------------------------------------------
#define BM 128
#define BN 128
#define BK 32
#define LPAD 8   // pad rows by 8 halves: keeps 16B alignment, spreads banks

__global__ __launch_bounds__(256)
void gemm_bias_kernel(const __bf16* __restrict__ A, const __bf16* __restrict__ Wt,
                      const float* __restrict__ bias,
                      const float* __restrict__ resid,
                      float* __restrict__ outf, __bf16* __restrict__ outb,
                      int M, int N, int K, int do_gelu) {
    __shared__ __align__(16) __bf16 As[2][BM][BK + LPAD];
    __shared__ __align__(16) __bf16 Ws[2][BN][BK + LPAD];

    const int tid  = threadIdx.x;
    const int wave = tid >> 5;
    const int lane = tid & 31;
    const int half = lane >> 4;
    const int l16  = lane & 15;
    const int wr   = wave & 3;    // 4 M-rows of 32
    const int wc   = wave >> 2;   // 2 N-cols of 64
    const int m0   = blockIdx.y * BM;
    const int n0   = blockIdx.x * BN;

    // cooperative copy mapping: 256 threads x 16B, twice (A and W)
    const int cr  = tid >> 1;           // 0..127
    const int cc  = (tid & 1) << 4;     // 0 or 16

    v8f acc[2][4] = {};

    const int ntiles = K / BK;

    // prefetch tile 0 into buffer 0
    async_copy16(&A [(size_t)(m0 + cr) * K + cc], &As[0][cr][cc]);
    async_copy16(&Wt[(size_t)(n0 + cr) * K + cc], &Ws[0][cr][cc]);

    for (int t = 0; t < ntiles; ++t) {
        const int cur = t & 1;
        if (t + 1 < ntiles) {
            const int k0 = (t + 1) * BK;
            async_copy16(&A [(size_t)(m0 + cr) * K + k0 + cc], &As[cur ^ 1][cr][cc]);
            async_copy16(&Wt[(size_t)(n0 + cr) * K + k0 + cc], &Ws[cur ^ 1][cr][cc]);
            // allow the 2 newest copies to stay in flight; tile `cur` is done
            wait_async_keep2();
        } else {
            wait_async_zero();
        }
        __syncthreads();

        v16bf a[2];
        #pragma unroll
        for (int mt = 0; mt < 2; ++mt)
            #pragma unroll
            for (int e = 0; e < 16; ++e)
                a[mt][e] = As[cur][wr * 32 + mt * 16 + l16][FRAG_K(e, half)];

        #pragma unroll
        for (int nt = 0; nt < 4; ++nt) {
            v16bf b;
            #pragma unroll
            for (int e = 0; e < 16; ++e)
                b[e] = Ws[cur][wc * 64 + nt * 16 + l16][FRAG_K(e, half)];
            acc[0][nt] = __builtin_amdgcn_wmma_f32_16x16x32_bf16(
                false, a[0], false, b, (short)0, acc[0][nt], false, false);
            acc[1][nt] = __builtin_amdgcn_wmma_f32_16x16x32_bf16(
                false, a[1], false, b, (short)0, acc[1][nt], false, false);
        }
        __syncthreads();
    }

    // epilogue: bias (+gelu) (+residual) -> f32 / bf16
    #pragma unroll
    for (int mt = 0; mt < 2; ++mt)
        #pragma unroll
        for (int nt = 0; nt < 4; ++nt) {
            const int n = n0 + wc * 64 + nt * 16 + l16;
            const float bb = bias ? bias[n] : 0.0f;
            #pragma unroll
            for (int j = 0; j < 8; ++j) {
                const int m = m0 + wr * 32 + mt * 16 + j + 8 * half;
                float v = acc[mt][nt][j] + bb;
                if (do_gelu) v = 0.5f * v * (1.0f + erff(v * 0.70710678118f));
                if (resid) v += resid[(size_t)m * N + n];
                if (outf) outf[(size_t)m * N + n] = v;
                if (outb) outb[(size_t)m * N + n] = (__bf16)v;
            }
        }
}

// ---------------------------------------------------------------------------
// Flash attention, one wave per (b, h, 16-row q-block).
// Mask causal|block_diag with TPB=16  <=>  key-block <= query-block, so the
// inner loop over 16-wide key blocks needs no element masking at all.
// ---------------------------------------------------------------------------
__global__ __launch_bounds__(32)
void attention_kernel(const __bf16* __restrict__ q, const __bf16* __restrict__ k,
                      const __bf16* __restrict__ v, __bf16* __restrict__ y) {
    __shared__ __bf16 Pl[16][34];   // P tile re-layout buffer (C-frag -> A-frag)

    const int lane = threadIdx.x;
    const int half = lane >> 4;
    const int l16  = lane & 15;
    const int qb   = blockIdx.x;    // 0..63 query block
    const int h    = blockIdx.y;
    const int b    = blockIdx.z;
    const size_t base = (size_t)b * SEQ * DMODEL + (size_t)h * DHEAD;

    // q A-fragments for DH=64 (two K=32 fragments)
    v16bf aq[2];
    {
        const int t = qb * 16 + l16;
        const __bf16* qp = q + base + (size_t)t * DMODEL;
        #pragma unroll
        for (int f = 0; f < 2; ++f)
            #pragma unroll
            for (int e = 0; e < 16; ++e)
                aq[f][e] = qp[f * 32 + FRAG_K(e, half)];
    }

    v8f o[4] = {};
    float m8[8], l8[8];
    #pragma unroll
    for (int j = 0; j < 8; ++j) { m8[j] = -1e30f; l8[j] = 0.0f; }

    const float scale = 0.125f;  // 1/sqrt(DH)

    for (int kb = 0; kb <= qb; kb += 2) {          // 32 keys per iteration
        const bool have2 = (kb + 1) <= qb;         // wave-uniform

        v8f s0 = {}, s1v = {};
        {
            const int s = kb * 16 + l16;
            const __bf16* kp = k + base + (size_t)s * DMODEL;
            #pragma unroll
            for (int f = 0; f < 2; ++f) {
                v16bf bk;
                #pragma unroll
                for (int e = 0; e < 16; ++e)
                    bk[e] = kp[f * 32 + FRAG_K(e, half)];
                s0 = __builtin_amdgcn_wmma_f32_16x16x32_bf16(
                    false, aq[f], false, bk, (short)0, s0, false, false);
            }
        }
        if (have2) {
            const int s = (kb + 1) * 16 + l16;
            const __bf16* kp = k + base + (size_t)s * DMODEL;
            #pragma unroll
            for (int f = 0; f < 2; ++f) {
                v16bf bk;
                #pragma unroll
                for (int e = 0; e < 16; ++e)
                    bk[e] = kp[f * 32 + FRAG_K(e, half)];
                s1v = __builtin_amdgcn_wmma_f32_16x16x32_bf16(
                    false, aq[f], false, bk, (short)0, s1v, false, false);
            }
        }

        // online softmax; C-fragment row m = j + 8*half lives in one 16-lane
        // group, so row reductions are xor-shuffles with masks 1,2,4,8
        #pragma unroll
        for (int j = 0; j < 8; ++j) {
            const float v0 = s0[j] * scale;
            const float v1 = have2 ? s1v[j] * scale : -1e30f;
            float cm = fmaxf(v0, v1);
            cm = fmaxf(cm, __shfl_xor(cm, 1, 32));
            cm = fmaxf(cm, __shfl_xor(cm, 2, 32));
            cm = fmaxf(cm, __shfl_xor(cm, 4, 32));
            cm = fmaxf(cm, __shfl_xor(cm, 8, 32));
            const float mn    = fmaxf(m8[j], cm);
            const float alpha = __expf(m8[j] - mn);
            const float p0 = __expf(v0 - mn);
            const float p1 = have2 ? __expf(v1 - mn) : 0.0f;
            float ps = p0 + p1;
            ps += __shfl_xor(ps, 1, 32);
            ps += __shfl_xor(ps, 2, 32);
            ps += __shfl_xor(ps, 4, 32);
            ps += __shfl_xor(ps, 8, 32);
            l8[j] = l8[j] * alpha + ps;
            m8[j] = mn;
            #pragma unroll
            for (int nt = 0; nt < 4; ++nt) o[nt][j] *= alpha;
            const int m = j + 8 * half;
            Pl[m][l16]      = (__bf16)p0;
            Pl[m][16 + l16] = (__bf16)p1;
        }
        __syncthreads();

        v16bf ap;
        #pragma unroll
        for (int e = 0; e < 16; ++e)
            ap[e] = Pl[l16][FRAG_K(e, half)];

        #pragma unroll
        for (int nt = 0; nt < 4; ++nt) {
            v16bf bvf;
            #pragma unroll
            for (int e = 0; e < 16; ++e) {
                int key = kb * 16 + FRAG_K(e, half);
                if (key >= (qb + 1) * 16) key = (qb + 1) * 16 - 1; // P=0 there
                bvf[e] = v[base + (size_t)key * DMODEL + nt * 16 + l16];
            }
            o[nt] = __builtin_amdgcn_wmma_f32_16x16x32_bf16(
                false, ap, false, bvf, (short)0, o[nt], false, false);
        }
        __syncthreads();
    }

    #pragma unroll
    for (int nt = 0; nt < 4; ++nt)
        #pragma unroll
        for (int j = 0; j < 8; ++j) {
            const int t = qb * 16 + j + 8 * half;
            const float val = o[nt][j] / l8[j];
            y[base + (size_t)t * DMODEL + nt * 16 + l16] = (__bf16)val;
        }
}

// ---------------------------------------------------------------------------
// Host-side orchestration
// ---------------------------------------------------------------------------
extern "C" void kernel_launch(void* const* d_in, const int* in_sizes, int n_in,
                              void* d_out, int out_size, void* d_ws, size_t ws_size,
                              hipStream_t stream) {
    (void)in_sizes; (void)n_in; (void)out_size; (void)ws_size;

    const float* seq   = (const float*)d_in[0];
    const float* Wq    = (const float*)d_in[1];
    const float* bq    = (const float*)d_in[2];
    const float* Wk    = (const float*)d_in[3];
    const float* bk    = (const float*)d_in[4];
    const float* Wv    = (const float*)d_in[5];
    const float* bv    = (const float*)d_in[6];
    const float* Wp    = (const float*)d_in[7];
    const float* bp    = (const float*)d_in[8];
    const float* ln1_g = (const float*)d_in[9];
    const float* ln1_b = (const float*)d_in[10];
    const float* ln2_g = (const float*)d_in[11];
    const float* ln2_b = (const float*)d_in[12];
    const float* W1    = (const float*)d_in[13];
    const float* b1    = (const float*)d_in[14];
    const float* W2    = (const float*)d_in[15];
    const float* b2    = (const float*)d_in[16];
    const float* lnf_g = (const float*)d_in[17];
    const float* lnf_b = (const float*)d_in[18];

    char* ws = (char*)d_ws;
    size_t off = 0;
    auto alloc = [&](size_t bytes) -> void* {
        void* p = ws + off;
        off += (bytes + 255) & ~(size_t)255;
        return p;
    };

    const size_t NWQ = (size_t)LAYERS * DMODEL * DMODEL;  // 8M elems
    const size_t NW1 = (size_t)LAYERS * DMODEL * DFF;     // 32M elems

    __bf16* wq_t = (__bf16*)alloc(NWQ * 2);   // all stored transposed [N][K]
    __bf16* wk_t = (__bf16*)alloc(NWQ * 2);
    __bf16* wv_t = (__bf16*)alloc(NWQ * 2);
    __bf16* wp_t = (__bf16*)alloc(NWQ * 2);
    __bf16* w1_t = (__bf16*)alloc(NW1 * 2);
    __bf16* w2_t = (__bf16*)alloc(NW1 * 2);
    float*  x    = (float*)alloc((size_t)ROWS * DMODEL * 4);
    __bf16* hbuf = (__bf16*)alloc((size_t)ROWS * DMODEL * 2);
    __bf16* qbuf = (__bf16*)alloc((size_t)ROWS * DMODEL * 2);
    __bf16* kbuf = (__bf16*)alloc((size_t)ROWS * DMODEL * 2);
    __bf16* vbuf = (__bf16*)alloc((size_t)ROWS * DMODEL * 2);
    __bf16* ybuf = (__bf16*)alloc((size_t)ROWS * DMODEL * 2);
    __bf16* gbuf = (__bf16*)alloc((size_t)ROWS * DFF * 2);

    // weights -> bf16, transposed to [N][K] (contiguous along K for async copies)
    const dim3 tb(32, 8);
    transpose_to_bf16_kernel<<<dim3(DMODEL/32, DMODEL/32, LAYERS), tb, 0, stream>>>(Wq, wq_t, DMODEL, DMODEL);
    transpose_to_bf16_kernel<<<dim3(DMODEL/32, DMODEL/32, LAYERS), tb, 0, stream>>>(Wk, wk_t, DMODEL, DMODEL);
    transpose_to_bf16_kernel<<<dim3(DMODEL/32, DMODEL/32, LAYERS), tb, 0, stream>>>(Wv, wv_t, DMODEL, DMODEL);
    transpose_to_bf16_kernel<<<dim3(DMODEL/32, DMODEL/32, LAYERS), tb, 0, stream>>>(Wp, wp_t, DMODEL, DMODEL);
    transpose_to_bf16_kernel<<<dim3(DFF/32,    DMODEL/32, LAYERS), tb, 0, stream>>>(W1, w1_t, DMODEL, DFF);
    transpose_to_bf16_kernel<<<dim3(DMODEL/32, DFF/32,    LAYERS), tb, 0, stream>>>(W2, w2_t, DFF, DMODEL);

    // residual stream x = sequences (f32 master copy)
    (void)hipMemcpyAsync(x, seq, (size_t)ROWS * DMODEL * 4,
                         hipMemcpyDeviceToDevice, stream);

    const dim3 gD(DMODEL / BN, ROWS / BM);  // (8, 32)
    const dim3 gF(DFF    / BN, ROWS / BM);  // (32, 32)
    const dim3 gA(SEQ / 16, NHEAD, BATCH);  // (64, 16, 4)

    for (int i = 0; i < LAYERS; ++i) {
        const size_t wo  = (size_t)i * DMODEL * DMODEL;
        const size_t wo1 = (size_t)i * DMODEL * DFF;

        layernorm_kernel<<<ROWS, 256, 0, stream>>>(
            x, ln1_g + i * DMODEL, ln1_b + i * DMODEL, nullptr, hbuf);
        gemm_bias_kernel<<<gD, 256, 0, stream>>>(
            hbuf, wq_t + wo, bq + i * DMODEL, nullptr, nullptr, qbuf,
            ROWS, DMODEL, DMODEL, 0);
        gemm_bias_kernel<<<gD, 256, 0, stream>>>(
            hbuf, wk_t + wo, bk + i * DMODEL, nullptr, nullptr, kbuf,
            ROWS, DMODEL, DMODEL, 0);
        gemm_bias_kernel<<<gD, 256, 0, stream>>>(
            hbuf, wv_t + wo, bv + i * DMODEL, nullptr, nullptr, vbuf,
            ROWS, DMODEL, DMODEL, 0);
        attention_kernel<<<gA, 32, 0, stream>>>(qbuf, kbuf, vbuf, ybuf);
        gemm_bias_kernel<<<gD, 256, 0, stream>>>(
            ybuf, wp_t + wo, bp + i * DMODEL, x, x, nullptr,
            ROWS, DMODEL, DMODEL, 0);
        layernorm_kernel<<<ROWS, 256, 0, stream>>>(
            x, ln2_g + i * DMODEL, ln2_b + i * DMODEL, nullptr, hbuf);
        gemm_bias_kernel<<<gF, 256, 0, stream>>>(
            hbuf, w1_t + wo1, b1 + i * DFF, nullptr, nullptr, gbuf,
            ROWS, DFF, DMODEL, 1);
        gemm_bias_kernel<<<gD, 256, 0, stream>>>(
            gbuf, w2_t + wo1, b2 + i * DMODEL, x, x, nullptr,
            ROWS, DMODEL, DFF, 0);
    }

    layernorm_kernel<<<ROWS, 256, 0, stream>>>(
        x, lnf_g, lnf_b, (float*)d_out, nullptr);
}
